// LzMultiHeadSelfAttention_38190849196227
// MI455X (gfx1250) — matrix operand order repaired and checked by reading, not compiled, not run on hardware
//
#include <hip/hip_runtime.h>
#include <hip/hip_bf16.h>

typedef __bf16 bf16_t;
typedef __bf16 v8bf  __attribute__((ext_vector_type(8)));
typedef __bf16 v16bf __attribute__((ext_vector_type(16)));
typedef float  v8f   __attribute__((ext_vector_type(8)));
typedef uint32_t v4u  __attribute__((ext_vector_type(4)));
typedef int      v8i32 __attribute__((ext_vector_type(8)));
typedef int      v4i32 __attribute__((ext_vector_type(4)));

#define WMMA_BF16(a, b, c) \
  __builtin_amdgcn_wmma_f32_16x16x32_bf16(false, (a), false, (b), (short)0, (c), false, false)

static constexpr int B_ = 8, S_ = 1024, D_ = 1024, H_ = 16, M_ = 64;

// Load a 16-element bf16 fragment as two 8-element (16B) chunks.
// gap = element distance between the chunks (16 for A-frags, 8 for contiguous B-frags).
static __device__ inline v16bf load_frag2x16(const bf16_t* base, int gap) {
  union { v16bf v; v8bf h[2]; } u;
  u.h[0] = *reinterpret_cast<const v8bf*>(base);
  u.h[1] = *reinterpret_cast<const v8bf*>(base + gap);
  return u.v;
}

// ---------------- TDM: 2D tile load Global -> LDS (data_size = 8B units) ----------------
// tile_d0/tile_d1: tile dims; tens_d0/tens_d1: tensor dims (OOB); stride0: row stride.
static __device__ __attribute__((always_inline)) void
tdm_load_2d_to_lds(uint32_t lds_byte_addr, uint64_t gaddr,
                   uint32_t tile_d0, uint32_t tile_d1,
                   uint32_t tens_d0, uint32_t tens_d1, uint64_t stride0) {
  uint32_t g0[4] = {0, 0, 0, 0};
  uint32_t g1[8] = {0, 0, 0, 0, 0, 0, 0, 0};
  g0[0] = 1u;                                                  // count=1 (valid), user mode
  g0[1] = lds_byte_addr;                                       // lds_addr   [63:32]
  g0[2] = (uint32_t)(gaddr & 0xffffffffu);                     // global_addr[95:64]
  g0[3] = (uint32_t)((gaddr >> 32) & 0x1ffffffu) | (2u << 30); // [120:96] + type=2 (image)
  g1[0] = (3u << 16);                                          // data_size=8B, wg_mask=0
  g1[1] |= (tens_d0 & 0xffffu) << 16;                          // tensor_dim0 [79:48]
  g1[2] |= (tens_d0 >> 16) & 0xffffu;
  g1[2] |= (tens_d1 & 0xffffu) << 16;                          // tensor_dim1 [111:80]
  g1[3] |= (tens_d1 >> 16) & 0xffffu;
  g1[3] |= (tile_d0 & 0xffffu) << 16;                          // tile_dim0   [127:112]
  g1[4] |= (tile_d1 & 0xffffu);                                // tile_dim1   [143:128]
  g1[5]  = (uint32_t)(stride0 & 0xffffffffu);                  // tensor_dim0_stride [207:160]
  g1[6]  = (uint32_t)((stride0 >> 32) & 0xffffu);
  v4u   G0 = { g0[0], g0[1], g0[2], g0[3] };
  v8i32 G1 = { (int)g1[0], (int)g1[1], (int)g1[2], (int)g1[3],
               (int)g1[4], (int)g1[5], (int)g1[6], (int)g1[7] };
  v4i32 Z4 = { 0, 0, 0, 0 };
#if __clang_major__ >= 23
  v8i32 Z8 = { 0, 0, 0, 0, 0, 0, 0, 0 };
  __builtin_amdgcn_tensor_load_to_lds(G0, G1, Z4, Z4, Z8, 0);
#else
  __builtin_amdgcn_tensor_load_to_lds(G0, G1, Z4, Z4, 0);
#endif
}

// ---------------- prep: q -> bf16, mask[b,s] = any(q != 0) ----------------
__global__ void __launch_bounds__(256)
k_prep_q(const float* __restrict__ q, bf16_t* __restrict__ qb, float* __restrict__ mask) {
  const int row = blockIdx.x;                 // b*S + s
  const float* src = q + (size_t)row * D_;
  bf16_t* dst = qb + (size_t)row * D_;
  __shared__ int any_nz;
  if (threadIdx.x == 0) any_nz = 0;
  __syncthreads();
  int local = 0;
  for (int i = threadIdx.x; i < D_; i += 256) {
    float v = src[i];
    local |= (v != 0.0f);
    dst[i] = (bf16_t)v;
  }
  if (local) atomicOr(&any_nz, 1);
  __syncthreads();
  if (threadIdx.x == 0) mask[row] = any_nz ? 1.0f : 0.0f;
}

// ---------------- pack W1 [H,D,m] into per-lane B-fragment layout ----------------
__global__ void __launch_bounds__(256)
k_pack_w1(const float* __restrict__ w1, bf16_t* __restrict__ w1p) {
  const int tid = blockIdx.x * 256 + threadIdx.x;     // 0 .. 2^20-1
  const int i  = tid & 15;
  const int L  = (tid >> 4) & 31;
  const int kc = (tid >> 9) & 31;
  const int nt = (tid >> 14) & 3;
  const int h  = tid >> 16;
  const int d  = kc * 32 + (i >> 3) * 16 + (L >> 4) * 8 + (i & 7);
  const int n  = nt * 16 + (L & 15);
  w1p[tid] = (bf16_t)w1[((size_t)h * D_ + d) * M_ + n];
}

// ---------------- pack W2 [D,D] into per-lane B-fragment layout ----------------
__global__ void __launch_bounds__(256)
k_pack_w2(const float* __restrict__ w2, bf16_t* __restrict__ w2p) {
  const int tid = blockIdx.x * 256 + threadIdx.x;     // 0 .. 2^20-1
  const int i  = tid & 15;
  const int L  = (tid >> 4) & 31;
  const int kc = (tid >> 9) & 31;
  const int nt = tid >> 14;                           // 0..63
  const int k  = kc * 32 + (i >> 3) * 16 + (L >> 4) * 8 + (i & 7);
  const int n  = nt * 16 + (L & 15);
  w2p[tid] = (bf16_t)w2[(size_t)k * D_ + n];
}

// ---------------- GEMM1: p[b,h] = qb[b] @ W1[h]  (TDM-staged B, 32 rows/wave) ----------
__global__ void __launch_bounds__(128)
k_gemm_p(const bf16_t* __restrict__ qb, const bf16_t* __restrict__ w1p,
         bf16_t* __restrict__ p) {
  const int bh = blockIdx.y, b = bh >> 4, h = bh & 15;
  const int lane = threadIdx.x & 31, wave = threadIdx.x >> 5;
  const int half = lane >> 4, lan16 = lane & 15;
  const int rb = blockIdx.x * 128 + wave * 32;        // 2 M-tiles per wave

  __shared__ __align__(16) bf16_t Bsm[2][4 * 8 * 512];   // double-buffered 32KB panels
  const bf16_t* gB = w1p + (size_t)(h * 4) * 32 * 512;   // this head's packed panel

  const bf16_t* arow0 = qb + (size_t)b * S_ * D_ + (size_t)(rb + lan16) * D_ + half * 8;
  const bf16_t* arow1 = arow0 + (size_t)16 * D_;
  v8f acc0[4] = { {}, {}, {}, {} };
  v8f acc1[4] = { {}, {}, {}, {} };

  if (wave == 0)
    tdm_load_2d_to_lds((uint32_t)(uintptr_t)&Bsm[0][0], (uint64_t)(uintptr_t)gB,
                       1024, 4, 4096, 4, 4096);
  for (int g = 0; g < 4; ++g) {
    if (wave == 0) {
      if (g < 3) {
        tdm_load_2d_to_lds((uint32_t)(uintptr_t)&Bsm[(g + 1) & 1][0],
                           (uint64_t)(uintptr_t)(gB + (size_t)(g + 1) * 8 * 512),
                           1024, 4, 4096, 4, 4096);
        __builtin_amdgcn_s_wait_tensorcnt(1);       // current buffer done, next in flight
      } else {
        __builtin_amdgcn_s_wait_tensorcnt(0);
      }
    }
    __syncthreads();
    const bf16_t* Bb = &Bsm[g & 1][0];
    #pragma unroll
    for (int kcl = 0; kcl < 8; ++kcl) {
      const int kc = g * 8 + kcl;
      v16bf af0 = load_frag2x16(arow0 + kc * 32, 16);
      v16bf af1 = load_frag2x16(arow1 + kc * 32, 16);
      __builtin_prefetch(arow0 + kc * 32 + 128, 0, 0);
      #pragma unroll
      for (int nt = 0; nt < 4; ++nt) {
        v16bf bfr = load_frag2x16(Bb + (nt * 8 + kcl) * 512 + lane * 16, 8);
        acc0[nt] = WMMA_BF16(af0, bfr, acc0[nt]);
        acc1[nt] = WMMA_BF16(af1, bfr, acc1[nt]);
      }
    }
    __syncthreads();
  }
  #pragma unroll
  for (int nt = 0; nt < 4; ++nt)
    #pragma unroll
    for (int r = 0; r < 8; ++r) {
      p[((size_t)bh * S_ + rb +      half * 8 + r) * M_ + nt * 16 + lan16] = (bf16_t)acc0[nt][r];
      p[((size_t)bh * S_ + rb + 16 + half * 8 + r) * M_ + nt * 16 + lan16] = (bf16_t)acc1[nt][r];
    }
}

// ---------------- fused flash attention: o[j] = sum_i exp(p_i.p_j) p_i / (rowsum + eps) ----
__global__ void __launch_bounds__(128)
k_attn(const bf16_t* __restrict__ p, const float* __restrict__ mask,
       bf16_t* __restrict__ obuf) {
  const int bh = blockIdx.y, b = bh >> 4, h = bh & 15;
  const int lane = threadIdx.x & 31, wave = threadIdx.x >> 5;
  const int half = lane >> 4, lan16 = lane & 15;
  const bf16_t* pbh = p + (size_t)bh * S_ * M_;
  const int jb = blockIdx.x * 64 + wave * 16;         // 16 j-rows per wave

  __shared__ __align__(16) bf16_t piT[M_ * 32];       // Pi transposed: [m][i_local]
  __shared__ __align__(16) bf16_t wsm[4 * 16 * 32];   // per-wave exp(scores): [j][i]
  bf16_t* wsw = wsm + wave * 16 * 32;

  v16bf aq[2];
  {
    const bf16_t* arow = pbh + (size_t)(jb + lan16) * M_ + half * 8;
    aq[0] = load_frag2x16(arow,      16);
    aq[1] = load_frag2x16(arow + 32, 16);
  }

  v8f oacc[4] = { {}, {}, {}, {} };
  float rowsum[8] = {0, 0, 0, 0, 0, 0, 0, 0};

  for (int ibase = 0; ibase < S_; ibase += 32) {
    {
      const int r  = threadIdx.x & 31;                // i_local
      const int cg = threadIdx.x >> 5;                // 16-col group
      union { v16bf v; v8bf hh[2]; } u;
      const bf16_t* src = pbh + (size_t)(ibase + r) * M_ + cg * 16;
      u.hh[0] = *reinterpret_cast<const v8bf*>(src);
      u.hh[1] = *reinterpret_cast<const v8bf*>(src + 8);
      #pragma unroll
      for (int e = 0; e < 16; ++e) piT[(cg * 16 + e) * 32 + r] = u.v[e];
    }
    __syncthreads();

    #pragma unroll
    for (int it = 0; it < 2; ++it) {
      v8f c = {};
      #pragma unroll
      for (int kc = 0; kc < 2; ++kc) {
        const bf16_t* bb = pbh + (size_t)(ibase + it * 16 + lan16) * M_ + kc * 32 + half * 8;
        v16bf bfr = load_frag2x16(bb, 16);
        c = WMMA_BF16(aq[kc], bfr, c);
      }
      #pragma unroll
      for (int r = 0; r < 8; ++r) {
        float w = __expf(c[r]);
        float s = w;
        s += __shfl_xor(s, 1, 32);
        s += __shfl_xor(s, 2, 32);
        s += __shfl_xor(s, 4, 32);
        s += __shfl_xor(s, 8, 32);                    // row-sum within 16-lane half
        rowsum[r] += s;
        wsw[(half * 8 + r) * 32 + it * 16 + lan16] = (bf16_t)w;
      }
    }

    {
      v16bf aw = load_frag2x16(wsw + lan16 * 32 + half * 8, 16);
      #pragma unroll
      for (int nt = 0; nt < 4; ++nt) {
        v16bf bfr = load_frag2x16(piT + (nt * 16 + lan16) * 32 + half * 8, 16);
        oacc[nt] = WMMA_BF16(aw, bfr, oacc[nt]);
      }
    }
    __syncthreads();
  }

  #pragma unroll
  for (int r = 0; r < 8; ++r) {
    const int j = jb + half * 8 + r;
    const float mk = mask[b * S_ + j];
    const float sc = mk / (rowsum[r] + 8.0f * 1e-7f);
    #pragma unroll
    for (int nt = 0; nt < 4; ++nt)
      obuf[((size_t)(b * S_ + j)) * D_ + h * M_ + nt * 16 + lan16] = (bf16_t)(oacc[nt][r] * sc);
  }
}

// ---------------- GEMM2: out = obuf @ W2  (TDM-staged B, 32 rows/wave, fp32 out) --------
__global__ void __launch_bounds__(128)
k_gemm_out(const bf16_t* __restrict__ obuf, const bf16_t* __restrict__ w2p,
           float* __restrict__ out) {
  const int lane = threadIdx.x & 31, wave = threadIdx.x >> 5;
  const int half = lane >> 4, lan16 = lane & 15;
  const int rb = blockIdx.x * 128 + wave * 32;
  const int cb = blockIdx.y * 64;

  __shared__ __align__(16) bf16_t Bsm[2][4 * 8 * 512];
  const bf16_t* gB = w2p + (size_t)(blockIdx.y * 4) * 32 * 512;

  const bf16_t* arow0 = obuf + (size_t)(rb + lan16) * D_ + half * 8;
  const bf16_t* arow1 = arow0 + (size_t)16 * D_;
  v8f acc0[4] = { {}, {}, {}, {} };
  v8f acc1[4] = { {}, {}, {}, {} };

  if (wave == 0)
    tdm_load_2d_to_lds((uint32_t)(uintptr_t)&Bsm[0][0], (uint64_t)(uintptr_t)gB,
                       1024, 4, 4096, 4, 4096);
  for (int g = 0; g < 4; ++g) {
    if (wave == 0) {
      if (g < 3) {
        tdm_load_2d_to_lds((uint32_t)(uintptr_t)&Bsm[(g + 1) & 1][0],
                           (uint64_t)(uintptr_t)(gB + (size_t)(g + 1) * 8 * 512),
                           1024, 4, 4096, 4, 4096);
        __builtin_amdgcn_s_wait_tensorcnt(1);
      } else {
        __builtin_amdgcn_s_wait_tensorcnt(0);
      }
    }
    __syncthreads();
    const bf16_t* Bb = &Bsm[g & 1][0];
    #pragma unroll
    for (int kcl = 0; kcl < 8; ++kcl) {
      const int kc = g * 8 + kcl;
      v16bf af0 = load_frag2x16(arow0 + kc * 32, 16);
      v16bf af1 = load_frag2x16(arow1 + kc * 32, 16);
      __builtin_prefetch(arow0 + kc * 32 + 128, 0, 0);
      #pragma unroll
      for (int nt = 0; nt < 4; ++nt) {
        v16bf bfr = load_frag2x16(Bb + (nt * 8 + kcl) * 512 + lane * 16, 8);
        acc0[nt] = WMMA_BF16(af0, bfr, acc0[nt]);
        acc1[nt] = WMMA_BF16(af1, bfr, acc1[nt]);
      }
    }
    __syncthreads();
  }
  #pragma unroll
  for (int nt = 0; nt < 4; ++nt)
    #pragma unroll
    for (int r = 0; r < 8; ++r) {
      out[(size_t)(rb +      half * 8 + r) * D_ + cb + nt * 16 + lan16] = acc0[nt][r];
      out[(size_t)(rb + 16 + half * 8 + r) * D_ + cb + nt * 16 + lan16] = acc1[nt][r];
    }
}

// ---------------------------------------------------------------------------------------
extern "C" void kernel_launch(void* const* d_in, const int* in_sizes, int n_in,
                              void* d_out, int out_size, void* d_ws, size_t ws_size,
                              hipStream_t stream) {
  const float* query = (const float*)d_in[0];   // [B,S,D]
  const float* W1    = (const float*)d_in[1];   // [H,D,m]
  const float* W2    = (const float*)d_in[2];   // [H*m,D]

  char* ws = (char*)d_ws;
  size_t off = 0;
  bf16_t* qb   = (bf16_t*)(ws + off); off += (size_t)B_ * S_ * D_ * 2;        // 16MB
  bf16_t* pbuf = (bf16_t*)(ws + off); off += (size_t)B_ * H_ * S_ * M_ * 2;   // 16MB
  bf16_t* obuf = (bf16_t*)(ws + off); off += (size_t)B_ * S_ * D_ * 2;        // 16MB
  bf16_t* w1p  = (bf16_t*)(ws + off); off += (size_t)H_ * D_ * M_ * 2;        // 2MB
  bf16_t* w2p  = (bf16_t*)(ws + off); off += (size_t)D_ * D_ * 2;             // 2MB
  float*  mask = (float*)(ws + off);  off += (size_t)B_ * S_ * 4;             // 32KB

  k_prep_q<<<B_ * S_, 256, 0, stream>>>(query, qb, mask);
  k_pack_w1<<<(H_ * D_ * M_) / 256, 256, 0, stream>>>(W1, w1p);
  k_pack_w2<<<(D_ * D_) / 256, 256, 0, stream>>>(W2, w2p);
  k_gemm_p<<<dim3(S_ / 128, B_ * H_), 128, 0, stream>>>(qb, w1p, pbuf);
  k_attn<<<dim3(S_ / 64, B_ * H_), 128, 0, stream>>>(pbuf, mask, obuf);
  k_gemm_out<<<dim3((B_ * S_) / 128, D_ / 64), 128, 0, stream>>>(obuf, w2p, (float*)d_out);
}
